// Gemma4TextModel_154618823079
// MI455X (gfx1250) — compile-verified, matrix operand order checked by reading, not run on hardware
//
#include <hip/hip_runtime.h>
#include <hip/hip_bf16.h>

// ---------------- model constants ----------------
#define L_   4
#define H_   2048
#define NH_  8
#define NKV_ 2
#define D_   256
#define I_   8192
#define T_   2048
#define WIN_ 512
#define G_   (NH_ / NKV_)
#define EPS_ 1e-6f

typedef __bf16 bf16;
typedef __attribute__((ext_vector_type(16))) __bf16 v16bf;
typedef __attribute__((ext_vector_type(8)))  __bf16 v8bf;
typedef __attribute__((ext_vector_type(8)))  float  v8f;
typedef __attribute__((ext_vector_type(4)))  float  v4f;
typedef __attribute__((ext_vector_type(4)))  int    v4i;

// Address-space-typed int4 pointers matching the async-copy builtin signature:
// (v4i addrspace(1)*, v4i addrspace(3)*, imm offset, imm cpol)
using gv4i = __attribute__((address_space(1))) v4i;
using lv4i = __attribute__((address_space(3))) v4i;

// Async global->LDS copy path (CDNA5). Guarded so compilation can't regress.
#if defined(__gfx1250__) && __has_builtin(__builtin_amdgcn_global_load_async_to_lds_b128)
#define ASYNC_COPY 1
#else
#define ASYNC_COPY 0
#endif

// 16-byte global -> LDS copy. Async (ASYNCcnt-tracked, no VGPR round trip)
// when available; falls back to a synchronous vector copy otherwise.
__device__ __forceinline__ void cp16(void* lds_dst, const void* gsrc) {
#if ASYNC_COPY
  __builtin_amdgcn_global_load_async_to_lds_b128(
      (gv4i*)(uintptr_t)gsrc,
      (lv4i*)(uint32_t)(uintptr_t)lds_dst, 0, 0);
#else
  *(v8bf*)lds_dst = *(const v8bf*)gsrc;
#endif
}

// Wait until at most N async copies are outstanding (ASYNCcnt decrements
// in order, so N == size of the newest batch keeps it in flight).
template <int N>
__device__ __forceinline__ void wait_async() {
#if ASYNC_COPY
#if __has_builtin(__builtin_amdgcn_s_wait_asynccnt)
  __builtin_amdgcn_s_wait_asynccnt(N);
#else
  asm volatile("s_wait_asynccnt %0" ::"n"(N) : "memory");
#endif
#endif
}

// ---------------- GEMM tiling ----------------
#define BM   128
#define BN   64
#define BK   32
#define SSTR 40   // padded LDS row stride (bf16 elems) to spread banks

// Load a 16x32 bf16 WMMA fragment from LDS following the CDNA5 16-bit
// A/B layout: lanes 0-15 hold K(0..7,16..23), lanes 16-31 hold K(8..15,24..31).
__device__ __forceinline__ v16bf frag_ld(const bf16* p) {
  union { v16bf v; v8bf h[2]; } u;
  u.h[0] = *(const v8bf*)(p);
  u.h[1] = *(const v8bf*)(p + 16);
  return u.v;
}

// C = A(bf16, MxK) * B(TB, NxK)^T ; f32 accumulate via v_wmma_f32_16x16x32_bf16.
// Double-buffered LDS staging: tile i+1 streams in (async) while tile i computes.
// If causal != 0, block tiles with no (t,s) satisfying s<=t && t-s<win are skipped.
template <typename TB>
__global__ __launch_bounds__(256) void k_gemm(
    const bf16* __restrict__ A, const TB* __restrict__ B, float* __restrict__ C,
    int M, int N, int K, int lda, int ldb, int ldc, int causal, int win) {
  __shared__ __align__(16) bf16 As[2][BM * SSTR];
  __shared__ __align__(16) bf16 Bs[2][BN * SSTR];
  const int tid = threadIdx.x;
  const int m0 = blockIdx.y * BM;
  const int n0 = blockIdx.x * BN;
  if (causal) {
    if (n0 >= m0 + BM) return;                 // fully above diagonal
    if (m0 - (n0 + BN - 1) >= win) return;     // fully outside sliding window
  }
  constexpr bool BF = !__is_same(TB, float);
  constexpr int AOPS = BF ? 3 : 2;             // async ops issued per thread per tile
  const int wave = tid >> 5, lane = tid & 31;
  const int wm = (wave >> 1) * 32, wn = (wave & 1) * 32;
  const int cl = lane & 15, hi = lane >> 4;

  v8f acc[2][2];
#pragma unroll
  for (int i = 0; i < 2; i++)
#pragma unroll
    for (int j = 0; j < 2; j++)
#pragma unroll
      for (int e = 0; e < 8; e++) acc[i][j][e] = 0.0f;

  auto stage = [&](int k0, int buf) {
    // A tile (128x32 bf16): 512 16-byte chunks / 256 threads
#pragma unroll
    for (int u = 0; u < 2; u++) {
      int un = tid + u * 256;
      int r = un >> 2, kc = (un & 3) * 8;
      cp16(&As[buf][r * SSTR + kc], &A[(size_t)(m0 + r) * lda + k0 + kc]);
    }
    // B tile (64x32)
    int r = tid >> 2, kc = (tid & 3) * 8;
    if constexpr (!BF) {
      // f32 weights: prefetch ahead, convert to bf16 while staging
      const float* src = &B[(size_t)(n0 + r) * ldb + k0 + kc];
      if (k0 + 2 * BK < K) __builtin_prefetch(src + 2 * BK, 0, 1);
      v4f f0 = *(const v4f*)src;
      v4f f1 = *(const v4f*)(src + 4);
      v8bf o;
#pragma unroll
      for (int e = 0; e < 4; e++) { o[e] = (bf16)f0[e]; o[e + 4] = (bf16)f1[e]; }
      *(v8bf*)&Bs[buf][r * SSTR + kc] = o;
    } else {
      cp16(&Bs[buf][r * SSTR + kc], &B[(size_t)(n0 + r) * ldb + k0 + kc]);
    }
  };

  const int nk = K / BK;
  stage(0, 0);
  for (int i = 0; i < nk; i++) {
    const int cur = i & 1;
    if (i + 1 < nk) {
      stage((i + 1) * BK, cur ^ 1);   // overlap next tile with this tile's WMMA
      wait_async<AOPS>();             // current tile done; next still in flight
    } else {
      wait_async<0>();
    }
    __syncthreads();

    const int kb = hi * 8;
    const bf16* as = As[cur];
    const bf16* bs = Bs[cur];
    v16bf a0 = frag_ld(&as[(wm + cl) * SSTR + kb]);
    v16bf a1 = frag_ld(&as[(wm + 16 + cl) * SSTR + kb]);
    v16bf b0 = frag_ld(&bs[(wn + cl) * SSTR + kb]);
    v16bf b1 = frag_ld(&bs[(wn + 16 + cl) * SSTR + kb]);

    acc[0][0] = __builtin_amdgcn_wmma_f32_16x16x32_bf16(false, a0, false, b0, (short)0, acc[0][0], false, false);
    acc[0][1] = __builtin_amdgcn_wmma_f32_16x16x32_bf16(false, a0, false, b1, (short)0, acc[0][1], false, false);
    acc[1][0] = __builtin_amdgcn_wmma_f32_16x16x32_bf16(false, a1, false, b0, (short)0, acc[1][0], false, false);
    acc[1][1] = __builtin_amdgcn_wmma_f32_16x16x32_bf16(false, a1, false, b1, (short)0, acc[1][1], false, false);
    __syncthreads();
  }

  // C layout: lanes 0-15 -> N=lane, M=vg ; lanes 16-31 -> N=lane-16, M=vg+8
#pragma unroll
  for (int mt = 0; mt < 2; mt++)
#pragma unroll
    for (int nt = 0; nt < 2; nt++) {
      int mbase = m0 + wm + mt * 16 + hi * 8;
      int nbase = n0 + wn + nt * 16 + cl;
#pragma unroll
      for (int vg = 0; vg < 8; vg++)
        C[(size_t)(mbase + vg) * ldc + nbase] = acc[mt][nt][vg];
    }
}

__device__ __forceinline__ float gelu_t(float x) {
  float x3 = x * x * x;
  return 0.5f * x * (1.0f + tanhf(0.7978845608028654f * (x + 0.044715f * x3)));
}

// Fused gate/up GEMM: O = bf16( gelu(A*Bg^T) * (A*Bu^T) ), A bf16, Bg/Bu f32.
__global__ __launch_bounds__(256) void k_gateup(
    const bf16* __restrict__ A, const float* __restrict__ Bg,
    const float* __restrict__ Bu, bf16* __restrict__ O,
    int K, int lda, int ldb, int ldc) {
  __shared__ __align__(16) bf16 As[2][BM * SSTR];
  __shared__ __align__(16) bf16 Bsg[2][BN * SSTR];
  __shared__ __align__(16) bf16 Bsu[2][BN * SSTR];
  const int tid = threadIdx.x;
  const int m0 = blockIdx.y * BM, n0 = blockIdx.x * BN;
  const int wave = tid >> 5, lane = tid & 31;
  const int wm = (wave >> 1) * 32, wn = (wave & 1) * 32;
  const int cl = lane & 15, hi = lane >> 4;

  v8f accG[2][2], accU[2][2];
#pragma unroll
  for (int i = 0; i < 2; i++)
#pragma unroll
    for (int j = 0; j < 2; j++)
#pragma unroll
      for (int e = 0; e < 8; e++) { accG[i][j][e] = 0.0f; accU[i][j][e] = 0.0f; }

  auto stage = [&](int k0, int buf) {
#pragma unroll
    for (int u = 0; u < 2; u++) {
      int un = tid + u * 256;
      int r = un >> 2, kc = (un & 3) * 8;
      cp16(&As[buf][r * SSTR + kc], &A[(size_t)(m0 + r) * lda + k0 + kc]);
    }
    int r = tid >> 2, kc = (tid & 3) * 8;
    const float* sg = &Bg[(size_t)(n0 + r) * ldb + k0 + kc];
    const float* su = &Bu[(size_t)(n0 + r) * ldb + k0 + kc];
    if (k0 + 2 * BK < K) {
      __builtin_prefetch(sg + 2 * BK, 0, 1);
      __builtin_prefetch(su + 2 * BK, 0, 1);
    }
    v4f g0 = *(const v4f*)sg, g1 = *(const v4f*)(sg + 4);
    v4f u0 = *(const v4f*)su, u1 = *(const v4f*)(su + 4);
    v8bf og, ou;
#pragma unroll
    for (int e = 0; e < 4; e++) {
      og[e] = (bf16)g0[e]; og[e + 4] = (bf16)g1[e];
      ou[e] = (bf16)u0[e]; ou[e + 4] = (bf16)u1[e];
    }
    *(v8bf*)&Bsg[buf][r * SSTR + kc] = og;
    *(v8bf*)&Bsu[buf][r * SSTR + kc] = ou;
  };

  const int nk = K / BK;
  stage(0, 0);
  for (int i = 0; i < nk; i++) {
    const int cur = i & 1;
    if (i + 1 < nk) { stage((i + 1) * BK, cur ^ 1); wait_async<2>(); }
    else            { wait_async<0>(); }
    __syncthreads();

    const int kb = hi * 8;
    v16bf a0 = frag_ld(&As[cur][(wm + cl) * SSTR + kb]);
    v16bf a1 = frag_ld(&As[cur][(wm + 16 + cl) * SSTR + kb]);
    v16bf g0 = frag_ld(&Bsg[cur][(wn + cl) * SSTR + kb]);
    v16bf g1 = frag_ld(&Bsg[cur][(wn + 16 + cl) * SSTR + kb]);
    v16bf u0 = frag_ld(&Bsu[cur][(wn + cl) * SSTR + kb]);
    v16bf u1 = frag_ld(&Bsu[cur][(wn + 16 + cl) * SSTR + kb]);

    accG[0][0] = __builtin_amdgcn_wmma_f32_16x16x32_bf16(false, a0, false, g0, (short)0, accG[0][0], false, false);
    accG[0][1] = __builtin_amdgcn_wmma_f32_16x16x32_bf16(false, a0, false, g1, (short)0, accG[0][1], false, false);
    accG[1][0] = __builtin_amdgcn_wmma_f32_16x16x32_bf16(false, a1, false, g0, (short)0, accG[1][0], false, false);
    accG[1][1] = __builtin_amdgcn_wmma_f32_16x16x32_bf16(false, a1, false, g1, (short)0, accG[1][1], false, false);
    accU[0][0] = __builtin_amdgcn_wmma_f32_16x16x32_bf16(false, a0, false, u0, (short)0, accU[0][0], false, false);
    accU[0][1] = __builtin_amdgcn_wmma_f32_16x16x32_bf16(false, a0, false, u1, (short)0, accU[0][1], false, false);
    accU[1][0] = __builtin_amdgcn_wmma_f32_16x16x32_bf16(false, a1, false, u0, (short)0, accU[1][0], false, false);
    accU[1][1] = __builtin_amdgcn_wmma_f32_16x16x32_bf16(false, a1, false, u1, (short)0, accU[1][1], false, false);
    __syncthreads();
  }

#pragma unroll
  for (int mt = 0; mt < 2; mt++)
#pragma unroll
    for (int nt = 0; nt < 2; nt++) {
      int mbase = m0 + wm + mt * 16 + hi * 8;
      int nbase = n0 + wn + nt * 16 + cl;
#pragma unroll
      for (int vg = 0; vg < 8; vg++)
        O[(size_t)(mbase + vg) * ldc + nbase] =
            (bf16)(gelu_t(accG[mt][nt][vg]) * accU[mt][nt][vg]);
    }
}

// ---------------- elementwise / norm kernels ----------------

__global__ void k_embed(const float* __restrict__ embed, const int* __restrict__ ids,
                        float* __restrict__ h) {
  int t = blockIdx.x;
  const float* src = embed + (size_t)ids[t] * H_;
  float* dst = h + (size_t)t * H_;
  const float sc = 45.25483399593904f;  // sqrt(2048)
  for (int i = threadIdx.x; i < H_; i += blockDim.x) dst[i] = src[i] * sc;
}

// out = bf16( rms(x_row) * w )
__global__ void k_rms_bf16(const float* __restrict__ x, const float* __restrict__ w,
                           bf16* __restrict__ out) {
  __shared__ float sb[256];
  int t = blockIdx.x, tid = threadIdx.x;
  const float* xr = x + (size_t)t * H_;
  float xl[8], ss = 0.0f;
#pragma unroll
  for (int i = 0; i < 8; i++) { float v = xr[tid + 256 * i]; xl[i] = v; ss += v * v; }
  sb[tid] = ss; __syncthreads();
  for (int s = 128; s > 0; s >>= 1) { if (tid < s) sb[tid] += sb[tid + s]; __syncthreads(); }
  float scale = rsqrtf(sb[0] / (float)H_ + EPS_);
  bf16* o = out + (size_t)t * H_;
#pragma unroll
  for (int i = 0; i < 8; i++) { int c = tid + 256 * i; o[c] = (bf16)(xl[i] * scale * w[c]); }
}

// h += rms(y_row) * w
__global__ void k_rms_res(float* __restrict__ h, const float* __restrict__ y,
                          const float* __restrict__ w) {
  __shared__ float sb[256];
  int t = blockIdx.x, tid = threadIdx.x;
  const float* yr = y + (size_t)t * H_;
  float yl[8], ss = 0.0f;
#pragma unroll
  for (int i = 0; i < 8; i++) { float v = yr[tid + 256 * i]; yl[i] = v; ss += v * v; }
  sb[tid] = ss; __syncthreads();
  for (int s = 128; s > 0; s >>= 1) { if (tid < s) sb[tid] += sb[tid + s]; __syncthreads(); }
  float scale = rsqrtf(sb[0] / (float)H_ + EPS_);
  float* hr = h + (size_t)t * H_;
#pragma unroll
  for (int i = 0; i < 8; i++) { int c = tid + 256 * i; hr[c] += yl[i] * scale * w[c]; }
}

// out_f32 = rms(h_row) * w  (final norm -> d_out)
__global__ void k_rms_out(const float* __restrict__ x, const float* __restrict__ w,
                          float* __restrict__ out) {
  __shared__ float sb[256];
  int t = blockIdx.x, tid = threadIdx.x;
  const float* xr = x + (size_t)t * H_;
  float xl[8], ss = 0.0f;
#pragma unroll
  for (int i = 0; i < 8; i++) { float v = xr[tid + 256 * i]; xl[i] = v; ss += v * v; }
  sb[tid] = ss; __syncthreads();
  for (int s = 128; s > 0; s >>= 1) { if (tid < s) sb[tid] += sb[tid + s]; __syncthreads(); }
  float scale = rsqrtf(sb[0] / (float)H_ + EPS_);
  float* o = out + (size_t)t * H_;
#pragma unroll
  for (int i = 0; i < 8; i++) { int c = tid + 256 * i; o[c] = xl[i] * scale * w[c]; }
}

// per-(t,head): rms over D with norm weight, then RoPE; out layout [head][T][D]
__global__ void k_qknorm_rope(const float* __restrict__ x, const float* __restrict__ w,
                              bf16* __restrict__ out, int nheads, float theta) {
  __shared__ float sb[256];
  __shared__ float buf[D_];
  int b = blockIdx.x;
  int t = b / nheads, hd = b % nheads;
  int tid = threadIdx.x;  // == D element
  float v = x[((size_t)t * nheads + hd) * D_ + tid];
  sb[tid] = v * v; __syncthreads();
  for (int s = 128; s > 0; s >>= 1) { if (tid < s) sb[tid] += sb[tid + s]; __syncthreads(); }
  float scale = rsqrtf(sb[0] / (float)D_ + EPS_);
  buf[tid] = v * scale * w[tid];
  __syncthreads();
  int i2 = tid & 127;
  float inv = __powf(theta, -2.0f * (float)i2 / (float)D_);
  float ang = (float)t * inv;
  float c = __cosf(ang), s2 = __sinf(ang);
  float r = (tid < 128) ? (buf[tid] * c - buf[tid + 128] * s2)
                        : (buf[tid] * c + buf[tid - 128] * s2);
  out[((size_t)hd * T_ + t) * D_ + tid] = (bf16)r;
}

// v: rms_noscale over D, store transposed [kv][D][T] for the P*V GEMM
__global__ void k_vnorm_t(const float* __restrict__ x, bf16* __restrict__ vt) {
  __shared__ float sb[256];
  int b = blockIdx.x;
  int t = b / NKV_, kv = b % NKV_;
  int tid = threadIdx.x;
  float v = x[((size_t)t * NKV_ + kv) * D_ + tid];
  sb[tid] = v * v; __syncthreads();
  for (int s = 128; s > 0; s >>= 1) { if (tid < s) sb[tid] += sb[tid + s]; __syncthreads(); }
  float scale = rsqrtf(sb[0] / (float)D_ + EPS_);
  vt[((size_t)kv * D_ + tid) * T_ + t] = (bf16)(v * scale);
}

// row softmax with causal / sliding mask, write P as bf16
__global__ void k_softmax(const float* __restrict__ S, bf16* __restrict__ P, int sliding) {
  __shared__ float sb[256];
  int t = blockIdx.x, tid = threadIdx.x;
  const float* sr = S + (size_t)t * T_;
  float vals[8], mx = -3.4028235e38f;
#pragma unroll
  for (int i = 0; i < 8; i++) {
    int s = tid + 256 * i;
    bool ok = (s <= t) && (!sliding || (t - s) < WIN_);
    float v = -3.4028235e38f;
    if (ok) v = sr[s];
    vals[i] = v;
    mx = fmaxf(mx, v);
  }
  sb[tid] = mx; __syncthreads();
  for (int s = 128; s > 0; s >>= 1) { if (tid < s) sb[tid] = fmaxf(sb[tid], sb[tid + s]); __syncthreads(); }
  mx = sb[0]; __syncthreads();
  float sum = 0.0f;
#pragma unroll
  for (int i = 0; i < 8; i++) { float e = __expf(vals[i] - mx); vals[i] = e; sum += e; }
  sb[tid] = sum; __syncthreads();
  for (int s = 128; s > 0; s >>= 1) { if (tid < s) sb[tid] += sb[tid + s]; __syncthreads(); }
  float inv = 1.0f / sb[0];
  bf16* pr = P + (size_t)t * T_;
#pragma unroll
  for (int i = 0; i < 8; i++) pr[tid + 256 * i] = (bf16)(vals[i] * inv);
}

__global__ void k_cvt_bf16(const float* __restrict__ x, bf16* __restrict__ y, int n) {
  int i = blockIdx.x * blockDim.x + threadIdx.x;
  if (i < n) y[i] = (bf16)x[i];
}

// ---------------- host orchestration ----------------
extern "C" void kernel_launch(void* const* d_in, const int* in_sizes, int n_in,
                              void* d_out, int out_size, void* d_ws, size_t ws_size,
                              hipStream_t stream) {
  (void)in_sizes; (void)n_in; (void)out_size; (void)ws_size;
  const float* embed    = (const float*)d_in[0];
  const float* q_w      = (const float*)d_in[1];
  const float* k_w      = (const float*)d_in[2];
  const float* v_w      = (const float*)d_in[3];
  const float* o_w      = (const float*)d_in[4];
  const float* q_norm_w = (const float*)d_in[5];
  const float* k_norm_w = (const float*)d_in[6];
  const float* ln1_w    = (const float*)d_in[7];
  const float* ln2_w    = (const float*)d_in[8];
  const float* ln3_w    = (const float*)d_in[9];
  const float* ln4_w    = (const float*)d_in[10];
  const float* gate_w   = (const float*)d_in[11];
  const float* up_w     = (const float*)d_in[12];
  const float* down_w   = (const float*)d_in[13];
  const float* final_w  = (const float*)d_in[14];
  const int*   ids      = (const int*)d_in[15];

  char* ws = (char*)d_ws;
  size_t off = 0;
  auto alloc = [&](size_t bytes) -> void* {
    void* p = ws + off;
    off += (bytes + 255) & ~(size_t)255;
    return p;
  };
  // total ~164 MB of workspace
  float* h      = (float*)alloc((size_t)T_ * H_ * 4);
  bf16*  xnb    = (bf16*) alloc((size_t)T_ * H_ * 2);
  float* qf     = (float*)alloc((size_t)T_ * NH_ * D_ * 4);
  float* kf     = (float*)alloc((size_t)T_ * NKV_ * D_ * 4);
  float* vf     = (float*)alloc((size_t)T_ * NKV_ * D_ * 4);
  bf16*  qb     = (bf16*) alloc((size_t)NH_ * T_ * D_ * 2);
  bf16*  kb     = (bf16*) alloc((size_t)NKV_ * T_ * D_ * 2);
  bf16*  vtb    = (bf16*) alloc((size_t)NKV_ * D_ * T_ * 2);
  float* scores = (float*)alloc((size_t)T_ * T_ * 4);
  bf16*  pb     = (bf16*) alloc((size_t)T_ * T_ * 2);
  float* attnf  = (float*)alloc((size_t)T_ * NH_ * D_ * 4);
  bf16*  attnb  = (bf16*) alloc((size_t)T_ * NH_ * D_ * 2);
  float* projf  = (float*)alloc((size_t)T_ * H_ * 4);
  bf16*  gub    = (bf16*) alloc((size_t)T_ * I_ * 2);

  const dim3 blk(256);
  const int BIG = 1 << 30;

  k_embed<<<T_, blk, 0, stream>>>(embed, ids, h);

  for (int l = 0; l < L_; l++) {
    const int   sliding = (l % 2 == 0);   // layer types: sliding, full, sliding, full
    const float theta   = sliding ? 10000.0f : 1000000.0f;

    // ---- attention ----
    k_rms_bf16<<<T_, blk, 0, stream>>>(h, ln1_w + (size_t)l * H_, xnb);
    k_gemm<float><<<dim3((NH_ * D_) / BN, T_ / BM), blk, 0, stream>>>(
        xnb, q_w + (size_t)l * NH_ * D_ * H_, qf, T_, NH_ * D_, H_, H_, H_, NH_ * D_, 0, BIG);
    k_gemm<float><<<dim3((NKV_ * D_) / BN, T_ / BM), blk, 0, stream>>>(
        xnb, k_w + (size_t)l * NKV_ * D_ * H_, kf, T_, NKV_ * D_, H_, H_, H_, NKV_ * D_, 0, BIG);
    k_gemm<float><<<dim3((NKV_ * D_) / BN, T_ / BM), blk, 0, stream>>>(
        xnb, v_w + (size_t)l * NKV_ * D_ * H_, vf, T_, NKV_ * D_, H_, H_, H_, NKV_ * D_, 0, BIG);

    k_qknorm_rope<<<T_ * NH_,  blk, 0, stream>>>(qf, q_norm_w + (size_t)l * D_, qb, NH_,  theta);
    k_qknorm_rope<<<T_ * NKV_, blk, 0, stream>>>(kf, k_norm_w + (size_t)l * D_, kb, NKV_, theta);
    k_vnorm_t<<<T_ * NKV_, blk, 0, stream>>>(vf, vtb);

    for (int hd = 0; hd < NH_; hd++) {
      const int kv = hd / G_;
      // S = Q_h K_kv^T  (band-skipped for sliding layers)
      k_gemm<bf16><<<dim3(T_ / BN, T_ / BM), blk, 0, stream>>>(
          qb + (size_t)hd * T_ * D_, kb + (size_t)kv * T_ * D_, scores,
          T_, T_, D_, D_, D_, T_, 1, sliding ? WIN_ : BIG);
      k_softmax<<<T_, blk, 0, stream>>>(scores, pb, sliding);
      // O_h = P V_kv  via transposed V
      k_gemm<bf16><<<dim3(D_ / BN, T_ / BM), blk, 0, stream>>>(
          pb, vtb + (size_t)kv * D_ * T_, attnf + (size_t)hd * D_,
          T_, D_, T_, T_, T_, NH_ * D_, 0, BIG);
    }

    k_cvt_bf16<<<(T_ * NH_ * D_) / 256, blk, 0, stream>>>(attnf, attnb, T_ * NH_ * D_);
    k_gemm<float><<<dim3(H_ / BN, T_ / BM), blk, 0, stream>>>(
        attnb, o_w + (size_t)l * H_ * NH_ * D_, projf, T_, H_, NH_ * D_,
        NH_ * D_, NH_ * D_, H_, 0, BIG);
    k_rms_res<<<T_, blk, 0, stream>>>(h, projf, ln2_w + (size_t)l * H_);

    // ---- MLP ----
    k_rms_bf16<<<T_, blk, 0, stream>>>(h, ln3_w + (size_t)l * H_, xnb);
    k_gateup<<<dim3(I_ / BN, T_ / BM), blk, 0, stream>>>(
        xnb, gate_w + (size_t)l * I_ * H_, up_w + (size_t)l * I_ * H_, gub,
        H_, H_, H_, I_);
    k_gemm<float><<<dim3(H_ / BN, T_ / BM), blk, 0, stream>>>(
        gub, down_w + (size_t)l * H_ * I_, projf, T_, H_, I_, I_, I_, H_, 0, BIG);
    k_rms_res<<<T_, blk, 0, stream>>>(h, projf, ln4_w + (size_t)l * H_);
  }

  k_rms_out<<<T_, blk, 0, stream>>>(h, final_w, (float*)d_out);
}
